// TransformerLanguageModel_38551626449458
// MI455X (gfx1250) — matrix operand order repaired and checked by reading, not compiled, not run on hardware
//
#include <hip/hip_runtime.h>
#include <hip/hip_bf16.h>

// ---------------- model constants ----------------
constexpr int Lc = 4, Bc = 4, Tc = 2048, Cc = 1024, Hc = 16, Fc = 4096, Vc = 32000;
constexpr int Dc = Cc / Hc;            // 64
constexpr int BT = Bc * Tc;            // 8192 tokens

// ---------------- WMMA fragment types ----------------
typedef __attribute__((ext_vector_type(16))) _Float16 v16h;
typedef __attribute__((ext_vector_type(8)))  float    v8f;

union Frag {
  v16h h;
  uint4 q[2];
};

// ---------------- CDNA5 async global->LDS copy (ASYNCcnt path) ----------------
#if defined(__gfx1250__) && __has_builtin(__builtin_amdgcn_global_load_async_to_lds_b128)
#define ASYNC_CP 1
typedef __attribute__((ext_vector_type(4))) int vi4;
typedef __attribute__((address_space(1))) vi4* gvi4_p;
typedef __attribute__((address_space(3))) vi4* lvi4_p;
__device__ __forceinline__ void async_cp16(const _Float16* g, _Float16* l) {
  __builtin_amdgcn_global_load_async_to_lds_b128((gvi4_p)g, (lvi4_p)l, 0, 0);
}
__device__ __forceinline__ void async_wait0() {
  asm volatile("s_wait_asynccnt 0x0" ::: "memory");
}
#else
#define ASYNC_CP 0
__device__ __forceinline__ void async_cp16(const _Float16* g, _Float16* l) {
  *(uint4*)l = *(const uint4*)g;
}
__device__ __forceinline__ void async_wait0() {}
#endif

__device__ __forceinline__ float rowmax16(float v) {
  #pragma unroll
  for (int m = 8; m >= 1; m >>= 1) v = fmaxf(v, __shfl_xor(v, m, 32));
  return v;
}
__device__ __forceinline__ float rowsum16(float v) {
  #pragma unroll
  for (int m = 8; m >= 1; m >>= 1) v += __shfl_xor(v, m, 32);
  return v;
}

// ---------------- embedding: x = tok_emb[idx] + pos_emb[t] ----------------
__global__ void embed_kernel(const int* __restrict__ idx, const float* __restrict__ tok,
                             const float* __restrict__ pos, float* __restrict__ x) {
  const int bt = blockIdx.x;
  const int t = bt % Tc;
  const int token = idx[bt];
  for (int c = threadIdx.x; c < Cc; c += blockDim.x)
    x[(size_t)bt * Cc + c] = tok[(size_t)token * Cc + c] + pos[(size_t)t * Cc + c];
}

// ---------------- layernorm (f16 out, optional f32 out) ----------------
__global__ __launch_bounds__(256) void ln_kernel(const float* __restrict__ x,
                                                 const float* __restrict__ g,
                                                 const float* __restrict__ bb,
                                                 _Float16* __restrict__ o16,
                                                 float* __restrict__ o32) {
  __shared__ float red[256];
  const int row = blockIdx.x, tid = threadIdx.x;
  const float* xr = x + (size_t)row * Cc;
  float s = 0.f;
  for (int c = tid; c < Cc; c += 256) s += xr[c];
  red[tid] = s; __syncthreads();
  for (int st = 128; st; st >>= 1) { if (tid < st) red[tid] += red[tid + st]; __syncthreads(); }
  const float mean = red[0] * (1.f / Cc);
  __syncthreads();
  s = 0.f;
  for (int c = tid; c < Cc; c += 256) { float d = xr[c] - mean; s += d * d; }
  red[tid] = s; __syncthreads();
  for (int st = 128; st; st >>= 1) { if (tid < st) red[tid] += red[tid + st]; __syncthreads(); }
  const float rstd = rsqrtf(red[0] * (1.f / Cc) + 1e-5f);
  for (int c = tid; c < Cc; c += 256) {
    float val = (xr[c] - mean) * rstd * g[c] + bb[c];
    o16[(size_t)row * Cc + c] = (_Float16)val;
    if (o32) o32[(size_t)row * Cc + c] = val;
  }
}

// ---------------- f32 (KxN) -> f16 transposed (NxK) weight conversion ----------------
__global__ __launch_bounds__(256) void conv_t_kernel(const float* __restrict__ W,
                                                     _Float16* __restrict__ Wt,
                                                     int K, int N) {
  __shared__ float tile[32][33];
  const int n0 = blockIdx.x * 32, k0 = blockIdx.y * 32;
  const int tx = threadIdx.x, ty = threadIdx.y;   // 32 x 8
  #pragma unroll
  for (int i = 0; i < 32; i += 8)
    tile[ty + i][tx] = W[(size_t)(k0 + ty + i) * N + n0 + tx];
  __syncthreads();
  #pragma unroll
  for (int i = 0; i < 32; i += 8)
    Wt[(size_t)(n0 + ty + i) * K + k0 + tx] = (_Float16)tile[tx][ty + i];
}

// ---------------- WMMA GEMM: C[M,N] = A[M,K] (f16) * Bt[N,K]^T (f16) + bias ----------------
// MODE 0: f16 out, +bias | MODE 1: f16 out, +bias, ReLU | MODE 2: f32 out, +bias, +residual
// Double-buffered LDS, async global->LDS staging overlapped with WMMA compute.
template <int MODE>
__global__ __launch_bounds__(256) void gemm_kernel(const _Float16* __restrict__ A,
                                                   const _Float16* __restrict__ Bt,
                                                   const float* __restrict__ bias,
                                                   const float* __restrict__ res,
                                                   _Float16* __restrict__ out16,
                                                   float* __restrict__ out32,
                                                   int M, int N, int K) {
  constexpr int LDA = 40;                 // 32 + 8 halves pad
  __shared__ _Float16 lA[2][128 * LDA];
  __shared__ _Float16 lB[2][128 * LDA];
  const int tid = threadIdx.x;
  const int wave = tid >> 5, lane = tid & 31;
  const int l16 = lane & 15, hi = lane >> 4;
  const int m0 = blockIdx.y * 128, n0 = blockIdx.x * 128;
  const int wm = (wave >> 2) * 64, wn = (wave & 3) * 32;
  // staging slots: 512 16B transfers per matrix per tile, 256 threads x 2
  const int srow = tid >> 2, scol = (tid & 3) * 8;      // slot it=0
  const int srow2 = (tid + 256) >> 2, scol2 = ((tid + 256) & 3) * 8;

  auto stage = [&](int buf, int k0) {
    async_cp16(A + (size_t)(m0 + srow) * K + k0 + scol,  &lA[buf][srow * LDA + scol]);
    async_cp16(Bt + (size_t)(n0 + srow) * K + k0 + scol, &lB[buf][srow * LDA + scol]);
    async_cp16(A + (size_t)(m0 + srow2) * K + k0 + scol2,  &lA[buf][srow2 * LDA + scol2]);
    async_cp16(Bt + (size_t)(n0 + srow2) * K + k0 + scol2, &lB[buf][srow2 * LDA + scol2]);
  };

  v8f acc[4][2] = {};
  stage(0, 0);
  async_wait0();
  __syncthreads();
  for (int k0 = 0; k0 < K; k0 += 32) {
    const int buf = (k0 >> 5) & 1;
    if (k0 + 32 < K) stage(buf ^ 1, k0 + 32);   // prefetch next tile (overlaps WMMAs)
    const _Float16* cA = lA[buf];
    const _Float16* cB = lB[buf];
    Frag bfr[2];
    #pragma unroll
    for (int j = 0; j < 2; ++j) {
      int n = wn + j * 16 + l16;
      bfr[j].q[0] = *(const uint4*)&cB[n * LDA + hi * 8];
      bfr[j].q[1] = *(const uint4*)&cB[n * LDA + 16 + hi * 8];
    }
    #pragma unroll
    for (int i = 0; i < 4; ++i) {
      Frag afr;
      int m = wm + i * 16 + l16;
      afr.q[0] = *(const uint4*)&cA[m * LDA + hi * 8];
      afr.q[1] = *(const uint4*)&cA[m * LDA + 16 + hi * 8];
      #pragma unroll
      for (int j = 0; j < 2; ++j)
        acc[i][j] = __builtin_amdgcn_wmma_f32_16x16x32_f16(
            false, afr.h, false, bfr[j].h, (short)0, acc[i][j], false, false);
    }
    async_wait0();
    __syncthreads();
  }
  #pragma unroll
  for (int i = 0; i < 4; ++i) {
    #pragma unroll
    for (int j = 0; j < 2; ++j) {
      const int gc = n0 + wn + j * 16 + l16;
      const float bv = bias[gc];
      #pragma unroll
      for (int r = 0; r < 8; ++r) {
        const int gr = m0 + wm + i * 16 + r + hi * 8;
        float v = acc[i][j][r] + bv;
        if constexpr (MODE == 1) v = fmaxf(v, 0.f);
        if constexpr (MODE == 2)
          out32[(size_t)gr * N + gc] = v + res[(size_t)gr * N + gc];
        else
          out16[(size_t)gr * N + gc] = (_Float16)v;
      }
    }
  }
}

// ---------------- flash attention (causal), WMMA QK^T and P*V ----------------
// grid: (T/128, B*H), 256 threads. Wave w owns 16 query rows.
__global__ __launch_bounds__(256) void attn_kernel(const _Float16* __restrict__ q,
                                                   const _Float16* __restrict__ k,
                                                   const _Float16* __restrict__ v,
                                                   _Float16* __restrict__ o) {
  __shared__ _Float16 lK[32 * 72];        // 32 keys x 64 d (pad 72)
  __shared__ _Float16 lV[64 * 40];        // V transposed: 64 d x 32 keys (pad 40)
  __shared__ _Float16 lP[8 * 16 * 40];    // per-wave P tile 16 x 32 (pad 40)
  const int tid = threadIdx.x, wave = tid >> 5, lane = tid & 31;
  const int l16 = lane & 15, hi = lane >> 4;
  const int bh = blockIdx.y, b = bh / Hc, h = bh % Hc;
  const int qb = blockIdx.x;
  const int q0 = qb * 128 + wave * 16;
  const size_t headoff = (size_t)h * Dc;

  Frag aq[2];
  {
    const _Float16* qrow = q + ((size_t)(b * Tc) + q0 + l16) * Cc + headoff;
    #pragma unroll
    for (int f = 0; f < 2; ++f) {
      aq[f].q[0] = *(const uint4*)(qrow + f * 32 + hi * 8);
      aq[f].q[1] = *(const uint4*)(qrow + f * 32 + 16 + hi * 8);
    }
  }
  float mrow[8], srow[8];
  #pragma unroll
  for (int r = 0; r < 8; ++r) { mrow[r] = -1e30f; srow[r] = 0.f; }
  v8f oacc[4] = {};
  _Float16* myP = lP + wave * (16 * 40);
  const int nkb = (qb + 1) * 4;           // causal: keys up to block's max row
  const int krr = tid >> 3, kcc = (tid & 7) * 8;

  for (int kb = 0; kb < nkb; ++kb) {
    const _Float16* kgp = k + ((size_t)(b * Tc) + kb * 32 + krr) * Cc + headoff + kcc;
    const _Float16* vgp = v + ((size_t)(b * Tc) + kb * 32 + krr) * Cc + headoff + kcc;
    // stage K block (async DMA): 32 rows x 64 halves
    async_cp16(kgp, &lK[krr * 72 + kcc]);
    // stage V transposed (needs lane shuffle -> manual)
    {
      uint4 raw = *(const uint4*)vgp;
      const _Float16* hp = (const _Float16*)&raw;
      #pragma unroll
      for (int u = 0; u < 8; ++u) lV[(kcc + u) * 40 + krr] = hp[u];
    }
    if (kb + 1 < nkb) {   // warm L2/WGP$ one block ahead
      __builtin_prefetch(kgp + (size_t)32 * Cc, 0, 3);
      __builtin_prefetch(vgp + (size_t)32 * Cc, 0, 3);
    }
    async_wait0();
    __syncthreads();

    v8f sc[2] = {};
    #pragma unroll
    for (int sub = 0; sub < 2; ++sub) {
      #pragma unroll
      for (int f = 0; f < 2; ++f) {
        Frag bk;
        int n = sub * 16 + l16;
        bk.q[0] = *(const uint4*)&lK[n * 72 + f * 32 + hi * 8];
        bk.q[1] = *(const uint4*)&lK[n * 72 + f * 32 + 16 + hi * 8];
        sc[sub] = __builtin_amdgcn_wmma_f32_16x16x32_f16(
            false, aq[f].h, false, bk.h, (short)0, sc[sub], false, false);
      }
    }
    // scale + causal mask
    #pragma unroll
    for (int sub = 0; sub < 2; ++sub) {
      const int kt = kb * 32 + sub * 16 + l16;
      #pragma unroll
      for (int r = 0; r < 8; ++r) {
        const int tr = q0 + r + hi * 8;
        float val = sc[sub][r] * 0.125f;          // 1/sqrt(64)
        sc[sub][r] = (kt > tr) ? -1e30f : val;
      }
    }
    // online softmax update
    #pragma unroll
    for (int r = 0; r < 8; ++r) {
      float rm = rowmax16(fmaxf(sc[0][r], sc[1][r]));
      float mnew = fmaxf(mrow[r], rm);
      float scale = __expf(mrow[r] - mnew);
      mrow[r] = mnew;
      srow[r] *= scale;
      #pragma unroll
      for (int dt = 0; dt < 4; ++dt) oacc[dt][r] *= scale;
      float psum = 0.f;
      #pragma unroll
      for (int sub = 0; sub < 2; ++sub) {
        float p = __expf(sc[sub][r] - mnew);
        psum += p;
        myP[(r + hi * 8) * 40 + sub * 16 + l16] = (_Float16)p;
      }
      srow[r] += rowsum16(psum);
    }
    // P (16x32) * V (32x64)
    Frag ap;
    ap.q[0] = *(const uint4*)&myP[l16 * 40 + hi * 8];
    ap.q[1] = *(const uint4*)&myP[l16 * 40 + 16 + hi * 8];
    #pragma unroll
    for (int dt = 0; dt < 4; ++dt) {
      Frag bv;
      bv.q[0] = *(const uint4*)&lV[(dt * 16 + l16) * 40 + hi * 8];
      bv.q[1] = *(const uint4*)&lV[(dt * 16 + l16) * 40 + 16 + hi * 8];
      oacc[dt] = __builtin_amdgcn_wmma_f32_16x16x32_f16(
          false, ap.h, false, bv.h, (short)0, oacc[dt], false, false);
    }
    __syncthreads();
  }
  #pragma unroll
  for (int r = 0; r < 8; ++r) {
    const float inv = 1.f / srow[r];
    const size_t rowbase = ((size_t)(b * Tc) + q0 + r + hi * 8) * Cc + headoff;
    #pragma unroll
    for (int dt = 0; dt < 4; ++dt)
      o[rowbase + dt * 16 + l16] = (_Float16)(oacc[dt][r] * inv);
  }
}

// ---------------- last-token logits: out[b,v] = xn[b,T-1,:] . Wout[:,v] + bout[v] ----------------
__global__ __launch_bounds__(256) void logits_kernel(const float* __restrict__ xn,
                                                     const float* __restrict__ Wout,
                                                     const float* __restrict__ bout,
                                                     float* __restrict__ out) {
  const int vv = blockIdx.x * 256 + threadIdx.x;
  if (vv >= Vc) return;
  float acc[Bc] = {0.f, 0.f, 0.f, 0.f};
  for (int kk = 0; kk < Cc; ++kk) {
    const float w = Wout[(size_t)kk * Vc + vv];
    #pragma unroll
    for (int b = 0; b < Bc; ++b)
      acc[b] += xn[((size_t)(b * Tc + Tc - 1)) * Cc + kk] * w;
  }
  #pragma unroll
  for (int b = 0; b < Bc; ++b) out[(size_t)b * Vc + vv] = acc[b] + bout[vv];
}

// ---------------- host orchestration ----------------
extern "C" void kernel_launch(void* const* d_in, const int* in_sizes, int n_in,
                              void* d_out, int out_size, void* d_ws, size_t ws_size,
                              hipStream_t stream) {
  (void)in_sizes; (void)n_in; (void)out_size; (void)ws_size;
  const int*   idx    = (const int*)  d_in[0];
  const float* tok    = (const float*)d_in[1];
  const float* pos    = (const float*)d_in[2];
  const float* Wq     = (const float*)d_in[3];
  const float* bq     = (const float*)d_in[4];
  const float* Wk     = (const float*)d_in[5];
  const float* bk     = (const float*)d_in[6];
  const float* Wv     = (const float*)d_in[7];
  const float* bv     = (const float*)d_in[8];
  const float* Wo     = (const float*)d_in[9];
  const float* bo     = (const float*)d_in[10];
  const float* ln1g   = (const float*)d_in[11];
  const float* ln1b   = (const float*)d_in[12];
  const float* W1     = (const float*)d_in[13];
  const float* b1     = (const float*)d_in[14];
  const float* W2     = (const float*)d_in[15];
  const float* b2     = (const float*)d_in[16];
  const float* ln2g   = (const float*)d_in[17];
  const float* ln2b   = (const float*)d_in[18];
  const float* lnfg   = (const float*)d_in[19];
  const float* lnfb   = (const float*)d_in[20];
  const float* Wout   = (const float*)d_in[21];
  const float* bout   = (const float*)d_in[22];

  char* ws = (char*)d_ws;
  size_t off = 0;
  auto alloc = [&](size_t bytes) -> void* {
    void* p = ws + off;
    off += (bytes + 255) & ~(size_t)255;
    return p;
  };
  float*    x     = (float*)   alloc((size_t)BT * Cc * 4);
  _Float16* h16   = (_Float16*)alloc((size_t)BT * Cc * 2);
  _Float16* q16   = (_Float16*)alloc((size_t)BT * Cc * 2);
  _Float16* k16   = (_Float16*)alloc((size_t)BT * Cc * 2);
  _Float16* v16   = (_Float16*)alloc((size_t)BT * Cc * 2);
  _Float16* a16   = (_Float16*)alloc((size_t)BT * Cc * 2);
  _Float16* mid16 = (_Float16*)alloc((size_t)BT * Fc * 2);
  float*    xn    = (float*)   alloc((size_t)BT * Cc * 4);
  _Float16* WqT[Lc]; _Float16* WkT[Lc]; _Float16* WvT[Lc]; _Float16* WoT[Lc];
  _Float16* W1T[Lc]; _Float16* W2T[Lc];
  for (int l = 0; l < Lc; ++l) {
    WqT[l] = (_Float16*)alloc((size_t)Cc * Cc * 2);
    WkT[l] = (_Float16*)alloc((size_t)Cc * Cc * 2);
    WvT[l] = (_Float16*)alloc((size_t)Cc * Cc * 2);
    WoT[l] = (_Float16*)alloc((size_t)Cc * Cc * 2);
    W1T[l] = (_Float16*)alloc((size_t)Cc * Fc * 2);
    W2T[l] = (_Float16*)alloc((size_t)Fc * Cc * 2);
  }

  const dim3 thr256(256), thrT(32, 8);
  // weight conversion (transposed f16)
  for (int l = 0; l < Lc; ++l) {
    const size_t wcc = (size_t)l * Cc * Cc;
    conv_t_kernel<<<dim3(Cc / 32, Cc / 32), thrT, 0, stream>>>(Wq + wcc, WqT[l], Cc, Cc);
    conv_t_kernel<<<dim3(Cc / 32, Cc / 32), thrT, 0, stream>>>(Wk + wcc, WkT[l], Cc, Cc);
    conv_t_kernel<<<dim3(Cc / 32, Cc / 32), thrT, 0, stream>>>(Wv + wcc, WvT[l], Cc, Cc);
    conv_t_kernel<<<dim3(Cc / 32, Cc / 32), thrT, 0, stream>>>(Wo + wcc, WoT[l], Cc, Cc);
    conv_t_kernel<<<dim3(Fc / 32, Cc / 32), thrT, 0, stream>>>(W1 + (size_t)l * Cc * Fc, W1T[l], Cc, Fc);
    conv_t_kernel<<<dim3(Cc / 32, Fc / 32), thrT, 0, stream>>>(W2 + (size_t)l * Fc * Cc, W2T[l], Fc, Cc);
  }

  embed_kernel<<<BT, thr256, 0, stream>>>(idx, tok, pos, x);

  const dim3 gCC(Cc / 128, BT / 128);   // N=1024
  const dim3 gCF(Fc / 128, BT / 128);   // N=4096
  const dim3 gAttn(Tc / 128, Bc * Hc);

  for (int l = 0; l < Lc; ++l) {
    const float* bql = bq + (size_t)l * Cc;
    const float* bkl = bk + (size_t)l * Cc;
    const float* bvl = bv + (size_t)l * Cc;
    const float* bol = bo + (size_t)l * Cc;
    ln_kernel<<<BT, thr256, 0, stream>>>(x, ln1g + (size_t)l * Cc, ln1b + (size_t)l * Cc, h16, nullptr);
    gemm_kernel<0><<<gCC, thr256, 0, stream>>>(h16, WqT[l], bql, nullptr, q16, nullptr, BT, Cc, Cc);
    gemm_kernel<0><<<gCC, thr256, 0, stream>>>(h16, WkT[l], bkl, nullptr, k16, nullptr, BT, Cc, Cc);
    gemm_kernel<0><<<gCC, thr256, 0, stream>>>(h16, WvT[l], bvl, nullptr, v16, nullptr, BT, Cc, Cc);
    attn_kernel<<<gAttn, thr256, 0, stream>>>(q16, k16, v16, a16);
    gemm_kernel<2><<<gCC, thr256, 0, stream>>>(a16, WoT[l], bol, x, nullptr, x, BT, Cc, Cc);
    ln_kernel<<<BT, thr256, 0, stream>>>(x, ln2g + (size_t)l * Cc, ln2b + (size_t)l * Cc, h16, nullptr);
    gemm_kernel<1><<<gCF, thr256, 0, stream>>>(h16, W1T[l], b1 + (size_t)l * Fc, nullptr, mid16, nullptr, BT, Fc, Cc);
    gemm_kernel<2><<<gCC, thr256, 0, stream>>>(mid16, W2T[l], b2 + (size_t)l * Cc, x, nullptr, x, BT, Cc, Fc);
  }

  ln_kernel<<<BT, thr256, 0, stream>>>(x, lnfg, lnfb, h16, xn);
  logits_kernel<<<(Vc + 255) / 256, thr256, 0, stream>>>(xn, Wout, bout, (float*)d_out);
}
